// AdaptiveEmbedding_7121055776930
// MI455X (gfx1250) — matrix operand order repaired and checked by reading, not compiled, not run on hardware
//
#include <hip/hip_runtime.h>

// ---------------------------------------------------------------------------
// AdaptiveEmbedding forward as three masked, accumulated GEMMs on WMMA bf16,
// software-pipelined with register-staged LDS double buffering.
//   tokens M = 8192, hidden N = 1024
//   cluster 0: K=1024, head_weight   (1024 x 20002) x head_weight_proj  (1024x1024)
//   cluster 1: K=256,  tail_weight_0 (256  x 60000) x tail_weight_proj_0 (1024x256)
//   cluster 2: K=64,   tail_weight_1 (64   x 120000) x tail_weight_proj_1 (1024x64)
//   y[t,h] = 32 * sum_k table_c[k, id_t - lo_c] * proj_c[h, k]
// Panel schedule: 32 + 8 + 2 = 42 K-panels of depth 32.
// ---------------------------------------------------------------------------

typedef __attribute__((ext_vector_type(16))) __bf16          v16bf;
typedef __attribute__((ext_vector_type(2)))  __bf16          v2bf;
typedef __attribute__((ext_vector_type(8)))  float           v8f;
typedef __attribute__((ext_vector_type(2)))  float           v2f;
typedef __attribute__((ext_vector_type(8)))  unsigned short  v8us;
typedef __attribute__((ext_vector_type(16))) unsigned short  v16us;

#define HID       1024
#define M_TOKENS  8192
#define MT        128      // block tile: tokens
#define NT        128      // block tile: hidden cols
#define AS        40       // LDS row stride in ushorts (80 B = 5*16 B aligned)
#define NPANEL    42       // 32 (c0) + 8 (c1) + 2 (c2)

// Pack two fp32 into two bf16 (RNE) in one dword; the vector conversion maps
// onto a single v_cvt_pk_bf16_f32 dst, a, b.
__device__ __forceinline__ unsigned int pack_bf16(float a, float b) {
  v2f  vf = {a, b};
  v2bf vb = __builtin_convertvector(vf, v2bf);
  return __builtin_bit_cast(unsigned int, vb);
}

__device__ __forceinline__ v16bf lds_frag(const unsigned short* p0,
                                          const unsigned short* p1) {
  v8us lo = *(const v8us*)p0;
  v8us hi = *(const v8us*)p1;
  v16us u = __builtin_shufflevector(lo, hi, 0, 1, 2, 3, 4, 5, 6, 7,
                                            8, 9, 10, 11, 12, 13, 14, 15);
  return __builtin_bit_cast(v16bf, u);
}

__global__ __launch_bounds__(256) void adaptive_embed_wmma(
    const int*   __restrict__ x,     // [8192]
    const float* __restrict__ hwp,   // [1024,1024]
    const float* __restrict__ hw,    // [1024,20002]
    const float* __restrict__ twp0,  // [1024,256]
    const float* __restrict__ tw0,   // [256,60000]
    const float* __restrict__ twp1,  // [1024,64]
    const float* __restrict__ tw1,   // [64,120000]
    float*       __restrict__ out)   // [8192,1024]
{
  __shared__ __attribute__((aligned(16))) unsigned short a_sh[2][MT * AS];
  __shared__ __attribute__((aligned(16))) unsigned short b_sh[2][NT * AS];

  const int tid  = threadIdx.x;
  const int lane = tid & 31;
  const int wid  = tid >> 5;
  const int wm   = wid & 3;            // 4 waves along M, 32 rows each
  const int wn   = wid >> 2;           // 2 waves along N, 64 cols each
  const int m0   = blockIdx.x * MT;
  const int n0   = blockIdx.y * NT;

  const int half = (lane < 16) ? 0 : 1;
  const int lr   = lane & 15;

  // --- A staging geometry: thread owns k-pair {2kp, 2kp+1} of the 32-wide
  //     panel for 8 rows {ra + 16*i}. Both gathers of a pair share one mask
  //     (single exec-predicated region) and one packed ds_store_b32.
  const int kp = tid & 15;             // k-pair index, k = 2kp, 2kp+1
  const int ra = tid >> 4;             // row base, rows ra + 16*i

  // Token ids are panel-invariant: keep them in VGPRs for the whole kernel.
  int ids[8];
#pragma unroll
  for (int i = 0; i < 8; ++i) ids[i] = x[m0 + ra + i * 16];

  // --- B staging geometry: float4 along k (proj rows are contiguous and
  //     16B-aligned for all K in {1024,256,64}).
  //     j = 0..3 : n = (tid + 256j)>>3, kq = ((tid + 256j)&7)*4

  v8f acc[2][4];
#pragma unroll
  for (int mt = 0; mt < 2; ++mt)
#pragma unroll
    for (int nt = 0; nt < 4; ++nt) acc[mt][nt] = (v8f){};

  float a0[8], a1[8];    // register-staged gathered embedding k-pair
  float4 b4[4];          // register-staged projection quads

  // ---- load panel p (global -> registers); gathers are exec-predicated so
  //      out-of-cluster rows cost no memory traffic, contribute exact zeros.
  auto load_panel = [&](int p) {
    const float* tab; const float* prj;
    int K, V, lo, hi, k0;
    if (p < 32)      { tab = hw;  prj = hwp;  K = 1024; V = 20002;  lo = 0;     hi = 20000;  k0 = p * 32; }
    else if (p < 40) { tab = tw0; prj = twp0; K = 256;  V = 60000;  lo = 20000; hi = 80000;  k0 = (p - 32) * 32; }
    else             { tab = tw1; prj = twp1; K = 64;   V = 120000; lo = 80000; hi = 200000; k0 = (p - 40) * 32; }
    const int kg = k0 + kp * 2;
#pragma unroll
    for (int i = 0; i < 8; ++i) {
      const int id = ids[i];
      float v0 = 0.0f, v1 = 0.0f;
      if (id >= lo && id < hi) {
        const float* base = tab + kg * V + (id - lo);
        v0 = base[0];
        v1 = base[V];
      }
      a0[i] = v0;
      a1[i] = v1;
    }
#pragma unroll
    for (int j = 0; j < 4; ++j) {
      const int e  = tid + j * 256;
      const int n  = e >> 3;
      const int kq = (e & 7) * 4;
      b4[j] = *(const float4*)&prj[(n0 + n) * K + k0 + kq];
    }
  };

  // ---- convert + store registers into LDS buffer b (packed stores)
  auto store_panel = [&](int b) {
    unsigned short* ap = &a_sh[b][0];
    unsigned short* bp = &b_sh[b][0];
#pragma unroll
    for (int i = 0; i < 8; ++i) {
      const int row = ra + i * 16;
      *(unsigned int*)&ap[row * AS + kp * 2] = pack_bf16(a0[i], a1[i]);
    }
#pragma unroll
    for (int j = 0; j < 4; ++j) {
      const int e  = tid + j * 256;
      const int n  = e >> 3;
      const int kq = (e & 7) * 4;
      *(uint2*)&bp[n * AS + kq] =
          make_uint2(pack_bf16(b4[j].x, b4[j].y), pack_bf16(b4[j].z, b4[j].w));
    }
  };

  // ---- 8 WMMAs on buffer b
  auto compute_panel = [&](int b) {
    v16bf afr[2];
#pragma unroll
    for (int mt = 0; mt < 2; ++mt) {
      const unsigned short* row = &a_sh[b][(wm * 32 + mt * 16 + lr) * AS];
      afr[mt] = lds_frag(row + (half ? 8 : 0), row + (half ? 24 : 16));
    }
#pragma unroll
    for (int nt = 0; nt < 4; ++nt) {
      const unsigned short* col =
          &b_sh[b][(wn * 64 + nt * 16 + lr) * AS + (half ? 8 : 0)];
      const v16bf bfr = lds_frag(col, col + 16);
#pragma unroll
      for (int mt = 0; mt < 2; ++mt) {
        acc[mt][nt] = __builtin_amdgcn_wmma_f32_16x16x32_bf16(
            false, afr[mt], false, bfr, (short)0, acc[mt][nt], false, false);
      }
    }
  };

  // ---- pipelined main loop: one barrier per panel; panel p+1's global loads
  //      issue while panel p's WMMAs run.
  load_panel(0);
  for (int p = 0; p < NPANEL; ++p) {
    const int b = p & 1;
    store_panel(b);
    __syncthreads();
    if (p + 1 < NPANEL) load_panel(p + 1);
    compute_panel(b);
  }

  // ---- writeback, scaled by sqrt(1024) = 32
#pragma unroll
  for (int mt = 0; mt < 2; ++mt) {
#pragma unroll
    for (int nt = 0; nt < 4; ++nt) {
      const int col = n0 + wn * 64 + nt * 16 + lr;
#pragma unroll
      for (int i = 0; i < 8; ++i) {
        const int row = m0 + wm * 32 + mt * 16 + (half ? (i + 8) : i);
        out[row * HID + col] = acc[mt][nt][i] * 32.0f;
      }
    }
  }
}

extern "C" void kernel_launch(void* const* d_in, const int* in_sizes, int n_in,
                              void* d_out, int out_size, void* d_ws, size_t ws_size,
                              hipStream_t stream) {
  (void)in_sizes; (void)n_in; (void)out_size; (void)d_ws; (void)ws_size;
  const int*   x    = (const int*)  d_in[0];
  const float* hwp  = (const float*)d_in[1];
  const float* hw   = (const float*)d_in[2];
  const float* twp0 = (const float*)d_in[3];
  const float* tw0  = (const float*)d_in[4];
  const float* twp1 = (const float*)d_in[5];
  const float* tw1  = (const float*)d_in[6];
  float* out = (float*)d_out;

  dim3 grid(M_TOKENS / MT, HID / NT);   // 64 x 8
  adaptive_embed_wmma<<<grid, 256, 0, stream>>>(x, hwp, hw, twp0, tw0, twp1, tw1, out);
}
